// SSM_18597208391915
// MI455X (gfx1250) — matrix operand order repaired and checked by reading, not compiled, not run on hardware
//
#include <hip/hip_runtime.h>
#include <math.h>

// Problem constants from the reference
#define HIDDEN 1024
#define SEQ    784
#define BATCH  512
#define OUTD   10
#define NBLK_X (BATCH / 32)     // 16  (M: 32 rows per single-wave workgroup)
#define NBLK_Y (HIDDEN / 64)    // 16  (N: 64 cols per single-wave workgroup)
#define NBLOCKS (NBLK_X * NBLK_Y)   // 256

typedef __attribute__((ext_vector_type(16))) __bf16       v16bf;
typedef __attribute__((ext_vector_type(2)))  __bf16       v2bf;
typedef __attribute__((ext_vector_type(8)))  float        v8f;
typedef __attribute__((ext_vector_type(8)))  unsigned int v8u;

// ---------- helpers ----------
__device__ __forceinline__ unsigned short f2bf(float f) {
    union { float f; unsigned int u; } c; c.f = f;
    unsigned int r = c.u + 0x7FFFu + ((c.u >> 16) & 1u);   // RNE
    return (unsigned short)(r >> 16);
}
__device__ __forceinline__ float bf2f(unsigned short v) {
    union { unsigned int u; float f; } c; c.u = ((unsigned int)v) << 16;
    return c.f;
}

#if __has_builtin(__builtin_amdgcn_tanhf)
__device__ __forceinline__ float fast_tanh(float x) { return __builtin_amdgcn_tanhf(x); } // v_tanh_f32
#else
__device__ __forceinline__ float fast_tanh(float x) { return tanhf(x); }
#endif

// store two bf16 results of one lane: lo -> p[0], hi -> p[ofs]
// Store pk elements directly: elem0 = b16 store of [15:0], elem1 = d16_hi store
// of [31:16] -- no integer extraction, no high-half zeroing movs.
__device__ __forceinline__ void store_bf16_pair(unsigned short* p, int ofs,
                                                float lo, float hi) {
#if __has_builtin(__builtin_amdgcn_cvt_pk_bf16_f32)
    v2bf pk = __builtin_amdgcn_cvt_pk_bf16_f32(lo, hi);     // v_cvt_pk_bf16_f32 (RNE)
    __bf16* q = (__bf16*)p;
    q[0]   = pk[0];
    q[ofs] = pk[1];
#else
    p[0]   = f2bf(lo);
    p[ofs] = f2bf(hi);
#endif
}

// A-operand (h) swizzle: 16-bit A 16x32 fragment layout (ISA 7.12.2).
__device__ __forceinline__ int awz(int m, int k) {
    return (((m >> 4) * 32 + (k >> 5)) * 512)
         + (((m & 15) | ((k & 8) << 1)) * 16)
         + (((k & 16) >> 1) | (k & 7));
}
// B-operand (AdT) swizzle: 16-bit B 32x16 fragment layout.
__device__ __forceinline__ int bwz(int k, int n) {
    return (((k >> 5) * 64 + (n >> 4)) * 512)
         + (((n & 15) | ((k & 8) << 1)) * 16)
         + (((k & 16) >> 1) | (k & 7));
}

// ---------- HiPPO setup ----------
// L = I - s*A, lower-triangular with rank-1 off-diagonal:
//   L[i][i] = 1 + s*(i+1),  L[i][j] = s*P_i*P_j (i>j),  P_i = sqrt(1+2i)
__global__ void hippo_invL(float* __restrict__ Mst) {
    int c = blockIdx.x * blockDim.x + threadIdx.x;
    if (c >= HIDDEN) return;
    const float s = 0.5f / (float)SEQ;
    for (int i = 0; i < c; ++i) Mst[i * HIDDEN + c] = 0.0f;
    float Pc = sqrtf(1.0f + 2.0f * (float)c);
    float xc = 1.0f / (1.0f + s * (float)(c + 1));
    Mst[c * HIDDEN + c] = xc;
    float S = Pc * xc;
    for (int i = c + 1; i < HIDDEN; ++i) {
        float Pi = sqrtf(1.0f + 2.0f * (float)i);
        float di = 1.0f + s * (float)(i + 1);
        float x  = -s * Pi * S / di;
        Mst[i * HIDDEN + c] = x;
        S += Pi * x;
    }
}

// Ad = M @ U, U = I + s*A:
//   Ad[i][k] = M[i][k]*(1 - s*(k+1)) - s*P_k * sum_{j>k} M[i][j]*P_j
__global__ void hippo_ad(const float* __restrict__ Mst,
                         unsigned short* __restrict__ Bwz,
                         float* __restrict__ Bvec) {
    int i = blockIdx.x * blockDim.x + threadIdx.x;
    if (i >= HIDDEN) return;
    const float s = 0.5f / (float)SEQ;
    float suffix = 0.0f;
    for (int k = HIDDEN - 1; k >= 0; --k) {
        float m   = Mst[i * HIDDEN + k];
        float Pk  = sqrtf(1.0f + 2.0f * (float)k);
        float ukk = 1.0f - s * (float)(k + 1);
        float ad  = m * ukk - s * Pk * suffix;   // Ad[i][k] == AdT[k][i]
        Bwz[bwz(k, i)] = f2bf(ad);
        suffix += m * Pk;
    }
    Bvec[i] = (1.0f / (float)SEQ) * suffix;
}

__global__ void zero_u32(unsigned int* __restrict__ p, int n) {
    int i = blockIdx.x * blockDim.x + threadIdx.x;
    if (i < n) p[i] = 0u;
}

// ---------- persistent scan kernel ----------
// One launch runs all 784 steps: h <- tanh(h @ AdT + u_t (x) Bvec).
// 256 single-wave workgroups (32 threads each): max spread across WGPs, and
// barrier ops degenerate to S_NOP (ISA: 1-wave groups). Device-wide split
// barrier between steps via a monotonically increasing counter
// (784*256 < 2^32): lane0 release-add, all lanes acquire-spin (wave-uniform).
__global__ __launch_bounds__(32)
void ssm_scan(unsigned short* __restrict__ hA,
              unsigned short* __restrict__ hB,
              const unsigned short* __restrict__ Bwz,
              const float* __restrict__ Bvec,
              const float* __restrict__ x,
              unsigned int* __restrict__ barrier_ctr) {
    const int lane  = threadIdx.x;          // 0..31
    const int waveM = blockIdx.x * 32;      // 32 batch rows
    const int waveN = blockIdx.y * 64;      // 64 hidden cols
    const int mtile0 = waveM >> 4;
    const int ntile0 = waveN >> 4;

    const int lhi = (lane >> 4) & 1;
    const int nl  = lane & 15;
    float bv[4];
#pragma unroll
    for (int nt = 0; nt < 4; ++nt) bv[nt] = Bvec[waveN + nt * 16 + nl];

    const v8f zero8 = {0.f, 0.f, 0.f, 0.f, 0.f, 0.f, 0.f, 0.f};

    for (int t = 0; t < SEQ; ++t) {
        const unsigned short* hin = (t & 1) ? hB : hA;
        unsigned short*       ho  = (t & 1) ? hA : hB;

        v8f acc[2][4];
#pragma unroll
        for (int mt = 0; mt < 2; ++mt)
#pragma unroll
            for (int nt = 0; nt < 4; ++nt) acc[mt][nt] = zero8;

        for (int kc = 0; kc < HIDDEN / 32; ++kc) {
            v16bf a[2], b[4];
#pragma unroll
            for (int mt = 0; mt < 2; ++mt) {
                const v8u* p = (const v8u*)(hin + ((mtile0 + mt) * 32 + kc) * 512 + lane * 16);
                a[mt] = __builtin_bit_cast(v16bf, *p);       // 2x global_load_b128
            }
#pragma unroll
            for (int nt = 0; nt < 4; ++nt) {
                const v8u* p = (const v8u*)(Bwz + (kc * 64 + ntile0 + nt) * 512 + lane * 16);
                b[nt] = __builtin_bit_cast(v16bf, *p);
            }
#pragma unroll
            for (int mt = 0; mt < 2; ++mt)
#pragma unroll
                for (int nt = 0; nt < 4; ++nt)
                    acc[mt][nt] = __builtin_amdgcn_wmma_f32_16x16x32_bf16(
                        false, a[mt], false, b[nt], (short)0, acc[mt][nt], false, false);
        }

        // epilogue (branch-free): element r of lane L -> (M = r + 8*(L>=16), N = L&15)
        // lane's 4 nt-results live at swizzled offsets base+{0, 8, 512, 520} (ushorts)
#pragma unroll
        for (int mt = 0; mt < 2; ++mt) {
#pragma unroll
            for (int r = 0; r < 8; ++r) {
                const int m = waveM + mt * 16 + r + lhi * 8;
                const float u = x[m * SEQ + t];              // x[b][t][0]
                float v0 = fast_tanh(acc[mt][0][r] + u * bv[0]);
                float v1 = fast_tanh(acc[mt][1][r] + u * bv[1]);
                float v2 = fast_tanh(acc[mt][2][r] + u * bv[2]);
                float v3 = fast_tanh(acc[mt][3][r] + u * bv[3]);
                unsigned short* pb = ho + awz(m, waveN + nl);
                store_bf16_pair(pb,       8, v0, v1);
                store_bf16_pair(pb + 512, 8, v2, v3);
            }
        }

        // ---- device-wide split barrier between steps ----
        __threadfence();                 // make h stores visible at device scope
        if (lane == 0)
            __hip_atomic_fetch_add(barrier_ctr, 1u, __ATOMIC_RELEASE,
                                   __HIP_MEMORY_SCOPE_AGENT);
        const unsigned target = (unsigned)(t + 1) * (unsigned)NBLOCKS;
        while (__hip_atomic_load(barrier_ctr, __ATOMIC_ACQUIRE,
                                 __HIP_MEMORY_SCOPE_AGENT) < target)
            __builtin_amdgcn_s_sleep(2);
    }
}

// ---------- readout: s[b] = h[b]·C (h read from swizzled bf16) ----------
__global__ void readout_dot(const unsigned short* __restrict__ hswz,
                            const float* __restrict__ C,
                            float* __restrict__ svec) {
    int wave = (blockIdx.x * blockDim.x + threadIdx.x) >> 5;
    int lane = threadIdx.x & 31;
    if (wave >= BATCH) return;
    float s = 0.0f;
    for (int j = lane; j < HIDDEN; j += 32)
        s += bf2f(hswz[awz(wave, j)]) * C[j];
#pragma unroll
    for (int off = 16; off > 0; off >>= 1)
        s += __shfl_xor(s, off, 32);
    if (lane == 0) svec[wave] = s;
}

__global__ void readout_out(const float* __restrict__ svec,
                            const float* __restrict__ W,
                            const float* __restrict__ bias,
                            float* __restrict__ out) {
    int i = blockIdx.x * blockDim.x + threadIdx.x;
    if (i >= BATCH * OUTD) return;
    int b = i / OUTD, k = i % OUTD;
    out[i] = svec[b] * W[k] + bias[k];
}

// ---------- launch ----------
extern "C" void kernel_launch(void* const* d_in, const int* in_sizes, int n_in,
                              void* d_out, int out_size, void* d_ws, size_t ws_size,
                              hipStream_t stream) {
    const float* x    = (const float*)d_in[0];   // (512,784,1)
    const float* C    = (const float*)d_in[1];   // (1,1024)
    const float* W    = (const float*)d_in[2];   // (1,10)
    const float* bias = (const float*)d_in[3];   // (10,)
    float* out = (float*)d_out;

    char* ws = (char*)d_ws;
    float*          Mst    = (float*)ws;          ws += (size_t)HIDDEN * HIDDEN * 4;
    unsigned short* Bwz    = (unsigned short*)ws; ws += (size_t)HIDDEN * HIDDEN * 2;
    float*          Bvec   = (float*)ws;          ws += (size_t)HIDDEN * 4;
    unsigned short* hA     = (unsigned short*)ws; ws += (size_t)BATCH * HIDDEN * 2;
    unsigned short* hB     = (unsigned short*)ws; ws += (size_t)BATCH * HIDDEN * 2;
    float*          svec   = (float*)ws;          ws += (size_t)BATCH * 4;
    unsigned int*   ctr    = (unsigned int*)ws;   ws += 64;   // barrier counter

    hippo_invL<<<HIDDEN / 256, 256, 0, stream>>>(Mst);
    hippo_ad<<<HIDDEN / 256, 256, 0, stream>>>(Mst, Bwz, Bvec);
    zero_u32<<<(BATCH * HIDDEN / 2 + 255) / 256, 256, 0, stream>>>((unsigned int*)hA,
                                                                   BATCH * HIDDEN / 2);
    zero_u32<<<1, 32, 0, stream>>>(ctr, 1);

    dim3 grid(NBLK_X, NBLK_Y), block(32);
    ssm_scan<<<grid, block, 0, stream>>>(hA, hB, Bwz, Bvec, x, ctr);

    // SEQ is even: step t = SEQ-1 (odd) wrote hA
    readout_dot<<<BATCH / 8, 256, 0, stream>>>(hA, C, svec);
    readout_out<<<(BATCH * OUTD + 255) / 256, 256, 0, stream>>>(svec, W, bias, out);
}
